// TimeMixing_38568806318348
// MI455X (gfx1250) — compile-verified, matrix operand order checked by reading, not run on hardware
//
#include <hip/hip_runtime.h>
#include <hip/hip_fp16.h>

typedef __attribute__((ext_vector_type(16))) _Float16     v16h;
typedef __attribute__((ext_vector_type(8)))  float        v8f;
typedef __attribute__((ext_vector_type(4)))  unsigned int v4u;

#define BB   64
#define SS   1024
#define DD   128
#define EPSV 1e-5f

// ---------------------------------------------------------------------------
// Kernel 1: per-batch mean / rsigma over S*D = 131072 elements
// ---------------------------------------------------------------------------
__global__ void tm_stats(const float* __restrict__ x, float* __restrict__ stats) {
    __shared__ float ssum[256];
    __shared__ float ssq[256];
    const int b = blockIdx.x;
    const float4* xb = (const float4*)(x + (size_t)b * SS * DD);
    float s = 0.f, q = 0.f;
    const int n4 = (SS * DD) / 4;
    for (int i = threadIdx.x; i < n4; i += 256) {
        float4 v = xb[i];
        s += v.x + v.y + v.z + v.w;
        q += v.x * v.x + v.y * v.y + v.z * v.z + v.w * v.w;
    }
    ssum[threadIdx.x] = s;
    ssq[threadIdx.x]  = q;
    __syncthreads();
    for (int st = 128; st > 0; st >>= 1) {
        if (threadIdx.x < st) {
            ssum[threadIdx.x] += ssum[threadIdx.x + st];
            ssq[threadIdx.x]  += ssq[threadIdx.x + st];
        }
        __syncthreads();
    }
    if (threadIdx.x == 0) {
        const float inv = 1.0f / (float)(SS * DD);
        float mu  = ssum[0] * inv;
        float var = ssq[0] * inv - mu * mu;
        stats[2 * b]     = mu;
        stats[2 * b + 1] = rsqrtf(var + EPSV);
    }
}

// ---------------------------------------------------------------------------
// Kernel 2: causal mask + f32 -> f16 convert of a (S,S) weight matrix
// ---------------------------------------------------------------------------
__global__ void tm_maskcvt(const float* __restrict__ W, _Float16* __restrict__ Wh) {
    const int idx = blockIdx.x * 256 + threadIdx.x;       // over S*S
    const int t = idx >> 10;          // / SS
    const int s = idx & (SS - 1);     // % SS
    Wh[idx] = (s <= t) ? (_Float16)W[idx] : (_Float16)0.0f;
}

// ---------------------------------------------------------------------------
// Kernel 3: layernorm + convert + transpose:  hT[b][d][s] (f16)
// ---------------------------------------------------------------------------
__global__ void tm_norm(const float* __restrict__ x,
                        const float* __restrict__ ln_w,
                        const float* __restrict__ ln_b,
                        const float* __restrict__ stats,
                        _Float16* __restrict__ hT) {
    const int idx = blockIdx.x * 256 + threadIdx.x;       // over B*S*D
    const int b   = idx / (SS * DD);
    const int rem = idx - b * (SS * DD);
    const int s   = rem >> 7;         // / DD
    const int d   = rem & (DD - 1);   // % DD
    const float mu = stats[2 * b];
    const float rs = stats[2 * b + 1];
    const float h  = (x[idx] - mu) * rs * ln_w[rem] + ln_b[rem];
    hT[(size_t)b * SS * DD + (size_t)d * SS + s] = (_Float16)h;
}

// ---------------------------------------------------------------------------
// WMMA GEMM, v3:
//   - workgroup (8 waves) -> 32(t) x 128(n) output block of one batch; wave w
//     owns n-tile w and both t-blocks, which share the causal K bound
//   - K-chunk = 64: 32x64 f16 A tile (4KB) staged to LDS with
//     global_load_async_to_lds_b128 (all 256 threads, one b128 each)
//   - 4-deep LDS ring + issue distance 2 + s_wait_asynccnt 2 => ONE barrier
//     per chunk (4 wmma / 8 ds_load_b128 / 4 global b128 per barrier); the
//     buffer overwritten at iter i was last read at iter i-2, separated by
//     the iter i-1 barrier
//   - global_prefetch on the per-wave B stream 4 chunks ahead
// ---------------------------------------------------------------------------
union ABPack { v16h v; v4u q[2]; unsigned int u[8]; };

__device__ __forceinline__ v8f tm_wmma(const v16h& a, const v16h& b, const v8f& c) {
    return __builtin_amdgcn_wmma_f32_16x16x32_f16(false, a, false, b,
                                                  (short)0, c, false, false);
}

template <bool HSWISH>
__global__ void tm_gemm(const _Float16* __restrict__ Wh,
                        const _Float16* __restrict__ Bt,   // (B, D, S) f16
                        const float* __restrict__ xres,    // residual (HSWISH=false)
                        void* __restrict__ outp) {
    __shared__ __align__(16) _Float16 sA[4][32 * 64];      // 4 x 4KB K-chunks

    const int tpair = blockIdx.x;            // 0..31 -> t rows [tpair*32, +32)
    const int b     = blockIdx.y;            // batch
    const int tid   = threadIdx.x;
    const int wave  = tid >> 5;              // 0..7 -> n-tile
    const int lane  = tid & 31;
    const int lo    = lane & 15;
    const int hi    = lane >> 4;
    const int trow  = tpair * 32;
    const int ncol  = wave * 16 + lo;
    const int nchunks = (tpair >> 1) + 1;    // causal bound rounded to 64

    // async producer: each thread moves one 16B segment of the 32x64 A tile
    const int arow = tid >> 3;               // 0..31
    const int aseg = tid & 7;                // 0..7
    const _Float16* agbase = Wh + (size_t)(trow + arow) * SS + aseg * 8;

    const _Float16* brow = Bt + (size_t)b * DD * SS + (size_t)ncol * SS + hi * 8;

    v8f c0 = {}, c1 = {};

#define TM_ISSUE(CH)                                                          \
    {                                                                         \
        unsigned ldsoff =                                                     \
            (unsigned)(size_t)&sA[(CH) & 3][arow * 64 + aseg * 8];            \
        const _Float16* g = agbase + (CH) * 64;                               \
        asm volatile("global_load_async_to_lds_b128 %0, %1, off"              \
                     :: "v"(ldsoff), "v"(g) : "memory");                      \
    }

    // prologue: chunks 0 and 1 in flight
    TM_ISSUE(0)
    if (nchunks > 1) TM_ISSUE(1)

    for (int i = 0; i < nchunks; ++i) {
        if (i + 2 < nchunks) {
            TM_ISSUE(i + 2)
            asm volatile("s_wait_asynccnt 0x2" ::: "memory");
        } else if (i + 1 < nchunks) {
            asm volatile("s_wait_asynccnt 0x1" ::: "memory");
        } else {
            asm volatile("s_wait_asynccnt 0x0" ::: "memory");
        }
        if (i + 4 < nchunks)
            __builtin_prefetch(brow + (i + 4) * 64, 0, 1);
        __syncthreads();                      // chunk i resident in LDS

        const _Float16* ab = &sA[i & 3][0];
#pragma unroll
        for (int s2 = 0; s2 < 2; ++s2) {      // two K=32 steps per chunk
            ABPack a0, a1, bm;
            const v4u* bp = (const v4u*)(brow + i * 64 + s2 * 32);
            bm.q[0] = bp[0];
            bm.q[1] = bp[2];
            const v4u* a0p = (const v4u*)(ab + lo * 64 + s2 * 32 + hi * 8);
            a0.q[0] = a0p[0];
            a0.q[1] = a0p[2];
            const v4u* a1p = (const v4u*)(ab + (16 + lo) * 64 + s2 * 32 + hi * 8);
            a1.q[0] = a1p[0];
            a1.q[1] = a1p[2];
            c0 = tm_wmma(a0.v, bm.v, c0);
            c1 = tm_wmma(a1.v, bm.v, c1);
        }
    }
#undef TM_ISSUE

    if (HSWISH) {
        _Float16* uT = (_Float16*)outp;       // (B, D, S) f16
        _Float16* ubase = uT + (size_t)b * DD * SS + (size_t)ncol * SS;
#pragma unroll
        for (int j = 0; j < 2; ++j) {
            const v8f& c = j ? c1 : c0;
            union { _Float16 h[8]; v4u u; } st;
#pragma unroll
            for (int r = 0; r < 8; ++r) {
                float uv = c[r];
                uv = uv * fminf(fmaxf(uv + 3.0f, 0.0f), 6.0f) * (1.0f / 6.0f);
                st.h[r] = (_Float16)uv;       // t = trow + j*16 + 8*hi + r
            }
            *(v4u*)(ubase + trow + j * 16 + hi * 8) = st.u;
        }
    } else {
        float* out = (float*)outp;            // (B, S, D) f32, += residual
#pragma unroll
        for (int j = 0; j < 2; ++j) {
            const v8f& c = j ? c1 : c0;
            const size_t base = (size_t)b * SS * DD +
                                (size_t)(trow + j * 16 + hi * 8) * DD + ncol;
#pragma unroll
            for (int r = 0; r < 8; ++r) {
                const size_t o = base + (size_t)r * DD;
                out[o] = xres[o] + c[r];
            }
        }
    }
}

// ---------------------------------------------------------------------------
extern "C" void kernel_launch(void* const* d_in, const int* in_sizes, int n_in,
                              void* d_out, int out_size, void* d_ws, size_t ws_size,
                              hipStream_t stream) {
    const float* x    = (const float*)d_in[0];
    const float* ln_w = (const float*)d_in[1];
    const float* ln_b = (const float*)d_in[2];
    const float* W1   = (const float*)d_in[3];
    const float* W2   = (const float*)d_in[4];
    float* out = (float*)d_out;

    char* ws = (char*)d_ws;
    float*    stats = (float*)ws;                                       // 512 B
    _Float16* W1h   = (_Float16*)(ws + 1024);                           // 2 MB
    _Float16* W2h   = (_Float16*)(ws + 1024 + (size_t)2 * 1024 * 1024); // 2 MB
    _Float16* hT    = (_Float16*)(ws + 1024 + (size_t)4 * 1024 * 1024); // 16 MB
    _Float16* uT    = (_Float16*)(ws + 1024 + (size_t)20 * 1024 * 1024);// 16 MB

    tm_stats<<<BB, 256, 0, stream>>>(x, stats);
    tm_maskcvt<<<(SS * SS) / 256, 256, 0, stream>>>(W1, W1h);
    tm_maskcvt<<<(SS * SS) / 256, 256, 0, stream>>>(W2, W2h);
    tm_norm<<<(BB * SS * DD) / 256, 256, 0, stream>>>(x, ln_w, ln_b, stats, hT);

    dim3 g(SS / 32, BB);
    tm_gemm<true ><<<g, 256, 0, stream>>>(W1h, hT, nullptr, uT);
    tm_gemm<false><<<g, 256, 0, stream>>>(W2h, uT, x, out);
}
